// RGNN_58841051955245
// MI455X (gfx1250) — matrix operand order repaired and checked by reading, not compiled, not run on hardware
//
#include <hip/hip_runtime.h>
#include <hip/hip_bf16.h>

// Problem constants (match reference)
#define NNODES 50000
#define MM     4
#define EDGES  400000
#define BGRAPH 64
#define HID    64
#define RTOT   (NNODES * MM)   // 200000 rows, divisible by 32

typedef __attribute__((ext_vector_type(16))) __bf16 v16bf;
typedef __attribute__((ext_vector_type(8)))  __bf16 v8bf;
typedef __attribute__((ext_vector_type(8)))  float  v8f;

// ---------------------------------------------------------------------------
// WMMA helpers: V_WMMA_F32_16X16X32_BF16, fragment layouts per ISA 7.12.2
// ---------------------------------------------------------------------------
static __device__ __forceinline__ v8f wmma_bf16(v16bf a, v16bf b, v8f c) {
  return __builtin_amdgcn_wmma_f32_16x16x32_bf16(
      /*neg_a=*/false, a, /*neg_b=*/false, b,
      /*c_mod=*/(short)0, c, /*reuse_a=*/false, /*reuse_b=*/false);
}

// A fragment (16x32, row-major source, ld = leading dim):
// lanes 0-15: M=lane, K = kt+{0..7, 16..23}; lanes 16-31: M=lane-16, K = kt+{8..15, 24..31}
// Two aligned 16-byte vector loads per lane.
static __device__ __forceinline__ v16bf load_a_bf16(const __bf16* __restrict__ X, int ld,
                                                    int rowTile, int kt, int lane) {
  int r  = rowTile + (lane & 15);
  int kb = kt + ((lane >> 4) << 3);
  const __bf16* p = X + (size_t)r * ld + kb;
  v8bf lo = *(const v8bf*)p;
  v8bf hi = *(const v8bf*)(p + 16);
  return __builtin_shufflevector(lo, hi, 0, 1, 2, 3, 4, 5, 6, 7,
                                         8, 9, 10, 11, 12, 13, 14, 15);
}

// Same A fragment from an f32 buffer (aggregation result): two 32-byte vector
// loads, convert in-register (v_cvt_pk_bf16_f32).
static __device__ __forceinline__ v16bf load_a_f32(const float* __restrict__ X, int ld,
                                                   int rowTile, int kt, int lane) {
  int r  = rowTile + (lane & 15);
  int kb = kt + ((lane >> 4) << 3);
  const float* p = X + (size_t)r * ld + kb;
  v8f lo = *(const v8f*)p;
  v8f hi = *(const v8f*)(p + 16);
  v16bf a;
#pragma unroll
  for (int j = 0; j < 8; ++j) a[j] = (__bf16)lo[j];
#pragma unroll
  for (int j = 0; j < 8; ++j) a[8 + j] = (__bf16)hi[j];
  return a;
}

// B fragments are pre-packed per (ktile, colTile, lane): one contiguous
// 32-byte load per lane per K-step.
static __device__ __forceinline__ v16bf load_b_packed(const __bf16* __restrict__ Wp,
                                                      int ktIdx, int colTileIdx, int lane) {
  const v16bf* p = (const v16bf*)(Wp + ((((size_t)ktIdx * 4 + colTileIdx) * 32 + lane) << 4));
  return *p;
}

// ---------------------------------------------------------------------------
// Weight packing: f32 W[K,64] row-major -> bf16 WMMA B-fragments.
// Fragment element j of lane L for (ktile, colTile):
//   K = ktile*32 + 8*(L>=16) + (j&7) + 16*(j>=8),  N = colTile*16 + (L&15)
// ---------------------------------------------------------------------------
__global__ void k_packw(const float* __restrict__ W, __bf16* __restrict__ out, int n) {
  int i = blockIdx.x * blockDim.x + threadIdx.x;   // n = K*64 exactly
  if (i >= n) return;
  int j    = i & 15;
  int lane = (i >> 4) & 31;
  int ct   = (i >> 9) & 3;
  int kt   = i >> 11;
  int k = kt * 32 + ((lane >> 4) << 3) + (j & 7) + ((j >> 3) << 4);
  int nn = ct * 16 + (lane & 15);
  out[i] = (__bf16)W[(size_t)k * HID + nn];
}

// ---------------------------------------------------------------------------
// Elementwise / scatter kernels
// ---------------------------------------------------------------------------
// agg0[dst, m] += x[src, m]   (Din == 1)
__global__ void k_agg0(const float* __restrict__ x, const int* __restrict__ ei,
                       float* __restrict__ agg0) {
  int e = blockIdx.x * blockDim.x + threadIdx.x;
  if (e >= EDGES) return;
  int s = ei[e];
  int d = ei[EDGES + e];
#pragma unroll
  for (int m = 0; m < MM; ++m)
    atomicAdd(&agg0[d * MM + m], x[s * MM + m]);
}

// Layer 0 conv: outer product (Din=1) + relu, emit bf16
__global__ void k_conv0(const float* __restrict__ x, const float* __restrict__ agg0,
                        const float* __restrict__ Ws, const float* __restrict__ Wn,
                        __bf16* __restrict__ X1) {
  int i = blockIdx.x * blockDim.x + threadIdx.x;   // RTOT*HID, exact
  int row = i >> 6, h = i & 63;
  float v = x[row] * Ws[h] + agg0[row] * Wn[h];
  X1[i] = (__bf16)(v > 0.f ? v : 0.f);
}

// Channel aggregation: AGG[dst,m,c] += X[src,m,c], one thread per (edge, c)
__global__ void k_aggc(const __bf16* __restrict__ X, const int* __restrict__ ei,
                       float* __restrict__ AGG) {
  int i = blockIdx.x * blockDim.x + threadIdx.x;   // EDGES*HID, exact
  int e = i >> 6, c = i & 63;
  int s = ei[e];
  int d = ei[EDGES + e];
#pragma unroll
  for (int m = 0; m < MM; ++m)
    atomicAdd(&AGG[(size_t)(d * MM + m) * HID + c], (float)X[(size_t)(s * MM + m) * HID + c]);
}

// ---------------------------------------------------------------------------
// WMMA GEMM kernels: block = 8 waves -> 32 rows x 64 cols of output
// ---------------------------------------------------------------------------
__global__ void k_conv_gemm(const __bf16* __restrict__ X, const float* __restrict__ AGG,
                            const __bf16* __restrict__ Wsp, const __bf16* __restrict__ Wnp,
                            __bf16* __restrict__ Y) {
  int lane = threadIdx.x & 31;
  int wave = threadIdx.x >> 5;
  int rowTile = blockIdx.x * 32 + (wave >> 2) * 16;
  int ct = wave & 3;
  int colTile = ct * 16;
  v8f c = {};
#pragma unroll
  for (int kt = 0; kt < HID; kt += 32) {
    c = wmma_bf16(load_a_bf16(X, HID, rowTile, kt, lane),
                  load_b_packed(Wsp, kt >> 5, ct, lane), c);
    c = wmma_bf16(load_a_f32(AGG, HID, rowTile, kt, lane),
                  load_b_packed(Wnp, kt >> 5, ct, lane), c);
  }
  int n  = colTile + (lane & 15);
  int rb = rowTile + ((lane >> 4) << 3);
#pragma unroll
  for (int r = 0; r < 8; ++r) {
    float v = c[r];
    Y[(size_t)(rb + r) * HID + n] = (__bf16)(v > 0.f ? v : 0.f);
  }
}

// MLP layer 1: K=192 across the three skip buffers (concat realized in the K loop)
__global__ void k_mlp1(const __bf16* __restrict__ X1, const __bf16* __restrict__ X2,
                       const __bf16* __restrict__ X3, const __bf16* __restrict__ W1p,
                       const float* __restrict__ b1, __bf16* __restrict__ H1) {
  int lane = threadIdx.x & 31;
  int wave = threadIdx.x >> 5;
  int rowTile = blockIdx.x * 32 + (wave >> 2) * 16;
  int ct = wave & 3;
  int colTile = ct * 16;
  const __bf16* srcs[3] = {X1, X2, X3};
  v8f c = {};
#pragma unroll
  for (int kt = 0; kt < 3 * HID; kt += 32) {
    const __bf16* S = srcs[kt / HID];
    c = wmma_bf16(load_a_bf16(S, HID, rowTile, kt & 63, lane),
                  load_b_packed(W1p, kt >> 5, ct, lane), c);
  }
  int n  = colTile + (lane & 15);
  int rb = rowTile + ((lane >> 4) << 3);
  float bias = b1[n];
#pragma unroll
  for (int r = 0; r < 8; ++r) {
    float v = c[r] + bias;
    H1[(size_t)(rb + r) * HID + n] = (__bf16)(v > 0.f ? v : 0.f);
  }
}

// MLP layer 2: 64->64, relu, bf16 out
__global__ void k_mlp2(const __bf16* __restrict__ H, const __bf16* __restrict__ Wp,
                       const float* __restrict__ bias, __bf16* __restrict__ Y) {
  int lane = threadIdx.x & 31;
  int wave = threadIdx.x >> 5;
  int rowTile = blockIdx.x * 32 + (wave >> 2) * 16;
  int ct = wave & 3;
  v8f c = {};
#pragma unroll
  for (int kt = 0; kt < HID; kt += 32)
    c = wmma_bf16(load_a_bf16(H, HID, rowTile, kt, lane),
                  load_b_packed(Wp, kt >> 5, ct, lane), c);
  int n  = ct * 16 + (lane & 15);
  int rb = rowTile + ((lane >> 4) << 3);
  float bv = bias[n];
#pragma unroll
  for (int r = 0; r < 8; ++r) {
    float v = c[r] + bv;
    Y[(size_t)(rb + r) * HID + n] = (__bf16)(v > 0.f ? v : 0.f);
  }
}

// MLP layer 3: 64->64, no relu, f32 out
__global__ void k_mlp3(const __bf16* __restrict__ H, const __bf16* __restrict__ Wp,
                       const float* __restrict__ bias, float* __restrict__ Y) {
  int lane = threadIdx.x & 31;
  int wave = threadIdx.x >> 5;
  int rowTile = blockIdx.x * 32 + (wave >> 2) * 16;
  int ct = wave & 3;
  v8f c = {};
#pragma unroll
  for (int kt = 0; kt < HID; kt += 32)
    c = wmma_bf16(load_a_bf16(H, HID, rowTile, kt, lane),
                  load_b_packed(Wp, kt >> 5, ct, lane), c);
  int n  = ct * 16 + (lane & 15);
  int rb = rowTile + ((lane >> 4) << 3);
  float bv = bias[n];
#pragma unroll
  for (int r = 0; r < 8; ++r)
    Y[(size_t)(rb + r) * HID + n] = c[r] + bv;
}

// ---------------------------------------------------------------------------
// Tail: mean over M, global min/max, scale + pool, final linear
// ---------------------------------------------------------------------------
__global__ void k_mean4(const float* __restrict__ H, float* __restrict__ G) {
  int i = blockIdx.x * blockDim.x + threadIdx.x;   // NNODES*HID, exact
  int nidx = i >> 6, c = i & 63;
  size_t base = (size_t)nidx * MM * HID + c;
  G[i] = 0.25f * (H[base] + H[base + HID] + H[base + 2 * HID] + H[base + 3 * HID]);
}

__global__ void k_minmax_partial(const float* __restrict__ g, float* __restrict__ pmin,
                                 float* __restrict__ pmax, int n) {
  __shared__ float smin[256], smax[256];
  float mn = 3.4e38f, mx = -3.4e38f;
  for (int i = blockIdx.x * blockDim.x + threadIdx.x; i < n; i += gridDim.x * blockDim.x) {
    float v = g[i];
    mn = fminf(mn, v);
    mx = fmaxf(mx, v);
  }
  int t = threadIdx.x;
  smin[t] = mn; smax[t] = mx;
  __syncthreads();
  for (int s = 128; s > 0; s >>= 1) {
    if (t < s) { smin[t] = fminf(smin[t], smin[t + s]); smax[t] = fmaxf(smax[t], smax[t + s]); }
    __syncthreads();
  }
  if (t == 0) { pmin[blockIdx.x] = smin[0]; pmax[blockIdx.x] = smax[0]; }
}

__global__ void k_minmax_final(const float* __restrict__ pmin, const float* __restrict__ pmax,
                               float* __restrict__ mm) {
  __shared__ float smin[256], smax[256];
  int t = threadIdx.x;
  smin[t] = pmin[t]; smax[t] = pmax[t];
  __syncthreads();
  for (int s = 128; s > 0; s >>= 1) {
    if (t < s) { smin[t] = fminf(smin[t], smin[t + s]); smax[t] = fmaxf(smax[t], smax[t + s]); }
    __syncthreads();
  }
  if (t == 0) { mm[0] = smin[0]; mm[1] = smax[0]; }
}

__global__ void k_scale_pool(const float* __restrict__ g, const int* __restrict__ batch,
                             const float* __restrict__ mm, float* __restrict__ pooled) {
  int i = blockIdx.x * blockDim.x + threadIdx.x;   // NNODES*HID, exact
  int nidx = i >> 6, c = i & 63;
  float mn = mm[0], mx = mm[1];
  float v = 2.f * (g[i] - mn) / (mx - mn);
  atomicAdd(&pooled[batch[nidx] * HID + c], v);
}

__global__ void k_out(const float* __restrict__ pooled, const float* __restrict__ Wo,
                      const float* __restrict__ bo, float* __restrict__ out) {
  int b = threadIdx.x;     // 64 threads
  float s = bo[0];
#pragma unroll
  for (int c = 0; c < HID; ++c) s += pooled[b * HID + c] * Wo[c];
  out[b] = s;
}

// ---------------------------------------------------------------------------
extern "C" void kernel_launch(void* const* d_in, const int* in_sizes, int n_in,
                              void* d_out, int out_size, void* d_ws, size_t ws_size,
                              hipStream_t stream) {
  (void)in_sizes; (void)n_in; (void)out_size; (void)ws_size;
  const float* x     = (const float*)d_in[0];
  const int*   ei    = (const int*)d_in[1];
  const int*   batch = (const int*)d_in[2];
  const float* Ws0   = (const float*)d_in[3];
  const float* Wn0   = (const float*)d_in[4];
  const float* Ws1   = (const float*)d_in[5];
  const float* Wn1   = (const float*)d_in[6];
  const float* Ws2   = (const float*)d_in[7];
  const float* Wn2   = (const float*)d_in[8];
  const float* W1    = (const float*)d_in[9];
  const float* b1    = (const float*)d_in[10];
  const float* W2    = (const float*)d_in[11];
  const float* b2    = (const float*)d_in[12];
  const float* W3    = (const float*)d_in[13];
  const float* b3    = (const float*)d_in[14];
  const float* Wo    = (const float*)d_in[15];
  const float* bo    = (const float*)d_in[16];
  float* out = (float*)d_out;

  // Workspace carve-up
  char* w = (char*)d_ws;
  auto carve = [&](size_t bytes) -> char* {
    char* p = w;
    w += (bytes + 255) & ~(size_t)255;
    return p;
  };
  const size_t szAct = (size_t)RTOT * HID * sizeof(__bf16);   // 25.6 MB
  __bf16* X1   = (__bf16*)carve(szAct);
  __bf16* X2   = (__bf16*)carve(szAct);
  __bf16* X3   = (__bf16*)carve(szAct);
  __bf16* H1   = (__bf16*)carve(szAct);
  __bf16* H2   = (__bf16*)carve(szAct);
  float*  AGG  = (float*)carve((size_t)RTOT * HID * sizeof(float)); // 51.2 MB; reused as MLP f32 out
  float*  agg0 = (float*)carve((size_t)RTOT * sizeof(float));
  float*  G    = (float*)carve((size_t)NNODES * HID * sizeof(float));
  __bf16* Ws1p = (__bf16*)carve(HID * HID * sizeof(__bf16));
  __bf16* Wn1p = (__bf16*)carve(HID * HID * sizeof(__bf16));
  __bf16* Ws2p = (__bf16*)carve(HID * HID * sizeof(__bf16));
  __bf16* Wn2p = (__bf16*)carve(HID * HID * sizeof(__bf16));
  __bf16* W1p  = (__bf16*)carve(3 * HID * HID * sizeof(__bf16));
  __bf16* W2p  = (__bf16*)carve(HID * HID * sizeof(__bf16));
  __bf16* W3p  = (__bf16*)carve(HID * HID * sizeof(__bf16));
  float*  pmin = (float*)carve(256 * sizeof(float));
  float*  pmax = (float*)carve(256 * sizeof(float));
  float*  mmx  = (float*)carve(2 * sizeof(float));
  float*  pooled = (float*)carve((size_t)BGRAPH * HID * sizeof(float));

  const int T = 256;

  // Pack weights into WMMA B-fragment layout (bf16)
  k_packw<<<(HID * HID + T - 1) / T, T, 0, stream>>>(Ws1, Ws1p, HID * HID);
  k_packw<<<(HID * HID + T - 1) / T, T, 0, stream>>>(Wn1, Wn1p, HID * HID);
  k_packw<<<(HID * HID + T - 1) / T, T, 0, stream>>>(Ws2, Ws2p, HID * HID);
  k_packw<<<(HID * HID + T - 1) / T, T, 0, stream>>>(Wn2, Wn2p, HID * HID);
  k_packw<<<(3 * HID * HID + T - 1) / T, T, 0, stream>>>(W1, W1p, 3 * HID * HID);
  k_packw<<<(HID * HID + T - 1) / T, T, 0, stream>>>(W2, W2p, HID * HID);
  k_packw<<<(HID * HID + T - 1) / T, T, 0, stream>>>(W3, W3p, HID * HID);

  // Conv layer 0 (Din = 1)
  hipMemsetAsync(agg0, 0, (size_t)RTOT * sizeof(float), stream);
  k_agg0<<<(EDGES + T - 1) / T, T, 0, stream>>>(x, ei, agg0);
  k_conv0<<<RTOT * HID / T, T, 0, stream>>>(x, agg0, Ws0, Wn0, X1);

  // Conv layer 1
  hipMemsetAsync(AGG, 0, (size_t)RTOT * HID * sizeof(float), stream);
  k_aggc<<<EDGES * HID / T, T, 0, stream>>>(X1, ei, AGG);
  k_conv_gemm<<<RTOT / 32, T, 0, stream>>>(X1, AGG, Ws1p, Wn1p, X2);

  // Conv layer 2
  hipMemsetAsync(AGG, 0, (size_t)RTOT * HID * sizeof(float), stream);
  k_aggc<<<EDGES * HID / T, T, 0, stream>>>(X2, ei, AGG);
  k_conv_gemm<<<RTOT / 32, T, 0, stream>>>(X2, AGG, Ws2p, Wn2p, X3);

  // MLP (K=192 concat realized inside k_mlp1)
  k_mlp1<<<RTOT / 32, T, 0, stream>>>(X1, X2, X3, W1p, b1, H1);
  k_mlp2<<<RTOT / 32, T, 0, stream>>>(H1, W2p, b2, H2);
  k_mlp3<<<RTOT / 32, T, 0, stream>>>(H2, W3p, b3, AGG);   // reuse AGG as f32 [RTOT, 64]

  // Mean over M, rescale, pool, output
  k_mean4<<<NNODES * HID / T, T, 0, stream>>>(AGG, G);
  k_minmax_partial<<<256, 256, 0, stream>>>(G, pmin, pmax, NNODES * HID);
  k_minmax_final<<<1, 256, 0, stream>>>(pmin, pmax, mmx);
  hipMemsetAsync(pooled, 0, (size_t)BGRAPH * HID * sizeof(float), stream);
  k_scale_pool<<<NNODES * HID / T, T, 0, stream>>>(G, batch, mmx, pooled);
  k_out<<<1, 64, 0, stream>>>(pooled, Wo, bo, out);
}
